// SWM_31190052503712
// MI455X (gfx1250) — compile-verified, matrix operand order checked by reading, not run on hardware
//
#include <hip/hip_runtime.h>

typedef __attribute__((ext_vector_type(16))) _Float16 v16h;
typedef __attribute__((ext_vector_type(8)))  _Float16 v8h;
typedef __attribute__((ext_vector_type(8)))  float    v8f;

// ---------- WMMA helpers (gfx1250 wave32, 16x16x32 f16 -> f32) ----------
// Fragment element map: lane-half h, element e -> K = (e>>3)*16 + (e&7) + 8h.
// => per lane, a fragment is two contiguous 8-half (16 B) runs at
//    K = k0 + 8h and K = k0 + 16 + 8h.  Load as two b128 + shuffle.
__device__ __forceinline__ v16h load_frag(const _Float16* row, int k0h) {
  const v8h lo = *(const v8h*)(row + k0h);
  const v8h hi = *(const v8h*)(row + k0h + 16);
  return __builtin_shufflevector(lo, hi, 0, 1, 2, 3, 4, 5, 6, 7, 8, 9, 10, 11,
                                 12, 13, 14, 15);
}

__device__ __forceinline__ v8f wmma16(v16h a, v16h b, v8f c) {
  return __builtin_amdgcn_wmma_f32_16x16x32_f16(false, a, false, b, (short)0, c,
                                                false, false);
}

// ---------- weight prep: f32 -> f16 (native row-major layouts) ----------
__global__ __launch_bounds__(256) void SWM_prep(
    const float* __restrict__ in_proj_w,   // (256,64)  -> W1h [n=256][k=64]
    const float* __restrict__ x_proj_w,    // (4,36,128)-> WAh [4][48(pad)][128]
    const float* __restrict__ out_proj_w,  // (64,128)  -> W3h [co=64][d=128]
    const float* __restrict__ A_logs,      // (512,16)  -> Aneg = -exp
    _Float16* __restrict__ W1h, _Float16* __restrict__ WAh,
    _Float16* __restrict__ W3h, float* __restrict__ Aneg)
{
  const int t0 = blockIdx.x * blockDim.x + threadIdx.x;
  const int stride = gridDim.x * blockDim.x;
  for (int i = t0; i < 256 * 64; i += stride) W1h[i] = (_Float16)in_proj_w[i];
  for (int i = t0; i < 4 * 48 * 128; i += stride) {
    int kdir = i / (48 * 128), rem = i % (48 * 128);
    int row = rem >> 7, kk = rem & 127;
    WAh[i] = (row < 36) ? (_Float16)x_proj_w[(kdir * 36 + row) * 128 + kk]
                        : (_Float16)0.0f;
  }
  for (int i = t0; i < 64 * 128; i += stride) W3h[i] = (_Float16)out_proj_w[i];
  for (int i = t0; i < 512 * 16; i += stride) Aneg[i] = -__expf(A_logs[i]);
}

// ---------- fused per-window SS2D kernel: one block per 8x8 window ----------
// LDS map (54912 B total):
//  R0 [0,17408):      s_u  f16 [64px][136d] conv+SiLU out (GEMM2 B rows, scan u)
//  R1 [17408,36992):  s_N  f16 [4][36][68]  x-proj results (scan B/C/dt)
//                     alias: s_xi f16 [64][136] (GEMM1 xi out, conv in)
//                     alias: s_g  f16 [64][136] (gated y, GEMM3 B rows)
//  R2 [36992,53888):  s_y  f16 [64][132]  direction-summed scan output
//                     alias: s_xc f16 [64][72] (window input, GEMM1 A rows)
//  R3 [53888,54400):  s_mu/s_rs f32 [64] (layernorm stats)
__global__ __launch_bounds__(256) void SWM_ss2d(
    const float* __restrict__ x, const float* __restrict__ pos,
    const float* __restrict__ conv_w, const float* __restrict__ conv_b,
    const float* __restrict__ dt_w, const float* __restrict__ dt_b,
    const float* __restrict__ Ds, const float* __restrict__ ln_w,
    const float* __restrict__ ln_b, const _Float16* __restrict__ W1h,
    const _Float16* __restrict__ WAh, const _Float16* __restrict__ W3h,
    const float* __restrict__ Aneg, float* __restrict__ out)
{
  __shared__ __attribute__((aligned(16))) char smem[54912];
  _Float16* s_u  = (_Float16*)(smem);
  _Float16* s_N  = (_Float16*)(smem + 17408);
  _Float16* s_xi = (_Float16*)(smem + 17408);
  _Float16* s_g  = (_Float16*)(smem + 17408);
  _Float16* s_y  = (_Float16*)(smem + 36992);
  _Float16* s_xc = (_Float16*)(smem + 36992);
  float*    s_mu = (float*)(smem + 53888);
  float*    s_rs = (float*)(smem + 54144);

  const int tid  = threadIdx.x;
  const int lane = tid & 31;
  const int l16  = lane & 15;
  const int half = (lane >> 4) & 1;
  const int k0h8 = 8 * half;
  const int wv   = tid >> 5;

  const int wid = blockIdx.x;      // 0..1023
  const int b   = wid >> 8;
  const int whh = (wid >> 4) & 15;
  const int www = wid & 15;

  v8h zch[8];  // odd waves: z tiles from GEMM1, kept in VGPRs until gating

  // ---- Phase A: gather rolled+scaled window, add pos -> s_xc [px][72] f16
  // lanes run along spatial (w-contiguous 32B global reads).
  for (int i = tid; i < 64 * 64; i += 256) {
    int px = i & 63, c = i >> 6;
    int r = px >> 3, cc = px & 7;
    int h = whh * 8 + r, w = www * 8 + cc;
    int hs = (h + 4) & 127, ws_ = (w + 4) & 127;   // roll(x, -4, -4)
    float v = x[((b * 64 + c) * 128 + hs) * 128 + ws_];
    bool sm = (w >= 124 && h < 120) || (h >= 124 && w < 120) ||
              (h >= 124 && w >= 124);
    v = sm ? v * 1e-4f : v;
    v += pos[(c * 8 + r) * 8 + cc];
    s_xc[px * 72 + c] = (_Float16)v;
  }
  __syncthreads();

  // ---- GEMM1 (WMMA): xz[64px][256] = xc[64][64] x in_proj^T
  // wave wv: M-tile = wv/2; even waves -> xi tiles (n 0..127) to LDS,
  // odd waves -> z tiles (n 128..255) stay in VGPRs.
  {
    const int mt = wv >> 1, m0 = mt * 16, mrow = m0 + l16;
    v8f acc[8];
#pragma unroll
    for (int j = 0; j < 8; j++)
      acc[j] = (v8f){0.f, 0.f, 0.f, 0.f, 0.f, 0.f, 0.f, 0.f};
#pragma unroll
    for (int ks = 0; ks < 2; ks++) {
      const int k0 = ks * 32;
      const v16h a = load_frag(s_xc + mrow * 72, k0 + k0h8);
#pragma unroll
      for (int j = 0; j < 8; j++) {
        const int n = ((wv & 1) * 8 + j) * 16 + l16;
        const v16h bf = load_frag(W1h + n * 64, k0 + k0h8);
        acc[j] = wmma16(a, bf, acc[j]);
      }
    }
    if ((wv & 1) == 0) {
#pragma unroll
      for (int j = 0; j < 8; j++) {
        const int n = j * 16 + l16;
#pragma unroll
        for (int v2 = 0; v2 < 8; v2++)
          s_xi[(m0 + v2 + 8 * half) * 136 + n] = (_Float16)acc[j][v2];
      }
    } else {
#pragma unroll
      for (int j = 0; j < 8; j++)
#pragma unroll
        for (int v2 = 0; v2 < 8; v2++) zch[j][v2] = (_Float16)acc[j][v2];
    }
  }
  __syncthreads();

  // ---- Phase B: depthwise 3x3 conv (pad 1) + SiLU -> s_u [px][136] f16
  {
    const int d = tid & 127;
    const int pxb = (tid >> 7) * 32;
    float w9[9];
#pragma unroll
    for (int i = 0; i < 9; i++) w9[i] = conv_w[d * 9 + i];
    const float bb = conv_b[d];
    for (int p = 0; p < 32; p++) {
      const int px = pxb + p, r = px >> 3, c = px & 7;
      float acc = bb;
#pragma unroll
      for (int dr = -1; dr <= 1; dr++)
#pragma unroll
        for (int dc = -1; dc <= 1; dc++) {
          int rr = r + dr, cc2 = c + dc;
          if (rr >= 0 && rr < 8 && cc2 >= 0 && cc2 < 8)
            acc += w9[(dr + 1) * 3 + dc + 1] *
                   (float)s_xi[(rr * 8 + cc2) * 136 + d];
        }
      s_u[px * 136 + d] = (_Float16)(acc / (1.0f + __expf(-acc)));  // SiLU
    }
  }
  __syncthreads();

  // ---- GEMM2 (WMMA): N_k[36][64] = x_proj_w[k][36][128] x u[128][64]
  // 4 dirs x 3 M-tiles x 4 N-tiles = 48 tiles, 6 per wave, K = 4x32.
  {
    for (int i = 0; i < 6; i++) {
      const int tt = wv * 6 + i;
      const int kdir = tt / 12, rem = tt % 12;
      const int m0 = (rem >> 2) * 16, n = (rem & 3) * 16 + l16;
      const int mrow = m0 + l16;
      v8f acc = (v8f){0.f, 0.f, 0.f, 0.f, 0.f, 0.f, 0.f, 0.f};
#pragma unroll
      for (int ks = 0; ks < 4; ks++) {
        const int k0 = ks * 32;
        const v16h a  = load_frag(WAh + (kdir * 48 + mrow) * 128, k0 + k0h8);
        const v16h bf = load_frag(s_u + n * 136, k0 + k0h8);
        acc = wmma16(a, bf, acc);
      }
#pragma unroll
      for (int v2 = 0; v2 < 8; v2++) {
        const int row = m0 + v2 + 8 * half;
        if (row < 36) s_N[(kdir * 36 + row) * 68 + n] = (_Float16)acc[v2];
      }
    }
  }
  __syncthreads();

  // ---- zero y accumulator
  {
    int* zp = (int*)s_y;
    for (int i = tid; i < (64 * 132) / 2; i += 256) zp[i] = 0;
  }
  __syncthreads();

  // ---- Phase D: selective scan. thread d (<128) owns channel d for all 4
  // directions sequentially (exclusive writer of column d of s_y -> no races).
  if (tid < 128) {
    const int d = tid;
    for (int kdir = 0; kdir < 4; kdir++) {
      float Ad[16];
#pragma unroll
      for (int n = 0; n < 16; n++) Ad[n] = Aneg[(kdir * 128 + d) * 16 + n];
      float dw0 = dt_w[(kdir * 128 + d) * 4 + 0];
      float dw1 = dt_w[(kdir * 128 + d) * 4 + 1];
      float dw2 = dt_w[(kdir * 128 + d) * 4 + 2];
      float dw3 = dt_w[(kdir * 128 + d) * 4 + 3];
      const float dtbv = dt_b[kdir * 128 + d];
      const float Dsv  = Ds[kdir * 128 + d];
      float hst[16];
#pragma unroll
      for (int n = 0; n < 16; n++) hst[n] = 0.f;
      for (int t = 0; t < 64; t++) {
        const int ttv = (kdir & 2) ? (63 - t) : t;
        const int s = (kdir & 1) ? (((ttv & 7) << 3) | (ttv >> 3)) : ttv;
        const _Float16* Ncol = &s_N[kdir * 36 * 68 + s];
        float dtraw = dtbv + dw0 * (float)Ncol[0] + dw1 * (float)Ncol[68] +
                      dw2 * (float)Ncol[136] + dw3 * (float)Ncol[204];
        const float delta =
            (dtraw > 20.f) ? dtraw : log1pf(__expf(dtraw));  // softplus
        const float u = (float)s_u[s * 136 + d];
        const float du = delta * u;
        float y = u * Dsv;
#pragma unroll
        for (int n = 0; n < 16; n++) {
          const float dA = __expf(delta * Ad[n]);
          hst[n] = hst[n] * dA + du * (float)Ncol[(4 + n) * 68];
          y += hst[n] * (float)Ncol[(20 + n) * 68];
        }
        _Float16* yp = &s_y[s * 132 + d];
        *yp = (_Float16)((float)*yp + y);
      }
    }
  }
  __syncthreads();

  // ---- layernorm stats per pixel
  if (tid < 64) {
    const int s = tid;
    float m = 0.f;
    for (int d = 0; d < 128; d++) m += (float)s_y[s * 132 + d];
    m *= (1.0f / 128.0f);
    float v = 0.f;
    for (int d = 0; d < 128; d++) {
      float df = (float)s_y[s * 132 + d] - m;
      v += df * df;
    }
    s_mu[s] = m;
    s_rs[s] = rsqrtf(v * (1.0f / 128.0f) + 1e-5f);
  }
  __syncthreads();

  // ---- gating: y_norm * SiLU(z); odd waves own z in VGPRs.
  // Write s_g [px][136] so GEMM3's B-fragment lanes (=pixels) read whole rows.
  if (wv & 1) {
    const int m0 = (wv >> 1) * 16;
#pragma unroll
    for (int j = 0; j < 8; j++) {
      const int ch = j * 16 + l16;
      const float lw = ln_w[ch], lb = ln_b[ch];
#pragma unroll
      for (int v2 = 0; v2 < 8; v2++) {
        const int px = m0 + v2 + 8 * half;
        const float yv = (float)s_y[px * 132 + ch];
        const float yn = (yv - s_mu[px]) * s_rs[px] * lw + lb;
        const float zv = (float)zch[j][v2];
        const float g = zv / (1.0f + __expf(-zv));
        s_g[px * 136 + ch] = (_Float16)(yn * g);
      }
    }
  }
  __syncthreads();

  // ---- GEMM3 (WMMA, transposed): out^T[64co][64px] = W3[64][128] x g^T
  // 16 tiles, 2 per wave; lanes = pixels -> 32B-contiguous global stores.
  {
    for (int i = 0; i < 2; i++) {
      const int tt = wv * 2 + i;
      const int m0 = (tt >> 2) * 16, n0 = (tt & 3) * 16;
      const int n = n0 + l16, mrow = m0 + l16;
      v8f acc = (v8f){0.f, 0.f, 0.f, 0.f, 0.f, 0.f, 0.f, 0.f};
#pragma unroll
      for (int ks = 0; ks < 4; ks++) {
        const int k0 = ks * 32;
        const v16h a  = load_frag(W3h + mrow * 128, k0 + k0h8);
        const v16h bf = load_frag(s_g + n * 136, k0 + k0h8);
        acc = wmma16(a, bf, acc);
      }
#pragma unroll
      for (int v2 = 0; v2 < 8; v2++) {
        const int co = m0 + v2 + 8 * half;
        const int px = n, r = px >> 3, c = px & 7;
        const int hf = (whh * 8 + r + 4) & 127;   // roll(y, +4, +4)
        const int wf = (www * 8 + c + 4) & 127;
        out[((b * 64 + co) * 128 + hf) * 128 + wf] = acc[v2];
      }
    }
  }
}

extern "C" void kernel_launch(void* const* d_in, const int* in_sizes, int n_in,
                              void* d_out, int out_size, void* d_ws,
                              size_t ws_size, hipStream_t stream) {
  const float* x          = (const float*)d_in[0];
  const float* pos        = (const float*)d_in[1];
  const float* in_proj_w  = (const float*)d_in[2];
  const float* conv_w     = (const float*)d_in[3];
  const float* conv_b     = (const float*)d_in[4];
  const float* x_proj_w   = (const float*)d_in[5];
  const float* dt_w       = (const float*)d_in[6];
  const float* dt_b       = (const float*)d_in[7];
  const float* A_logs     = (const float*)d_in[8];
  const float* Ds         = (const float*)d_in[9];
  const float* ln_w       = (const float*)d_in[10];
  const float* ln_b       = (const float*)d_in[11];
  const float* out_proj_w = (const float*)d_in[12];
  float* out = (float*)d_out;

  char* ws = (char*)d_ws;
  _Float16* W1h = (_Float16*)(ws);                    // 32768 B
  _Float16* WAh = (_Float16*)(ws + 32768);            // 49152 B
  _Float16* W3h = (_Float16*)(ws + 32768 + 49152);    // 16384 B
  float*    Aneg = (float*)(ws + 98304);              // 32768 B

  SWM_prep<<<64, 256, 0, stream>>>(in_proj_w, x_proj_w, out_proj_w, A_logs,
                                   W1h, WAh, W3h, Aneg);
  SWM_ss2d<<<1024, 256, 0, stream>>>(x, pos, conv_w, conv_b, dt_w, dt_b, Ds,
                                     ln_w, ln_b, W1h, WAh, W3h, Aneg, out);
}
